// StageNet_27410481283806
// MI455X (gfx1250) — matrix operand order repaired and checked by reading, not compiled, not run on hardware
//
#include <hip/hip_runtime.h>
#include <hip/hip_bf16.h>

// ---------------------------------------------------------------------------
// StageNet / ON-LSTM recurrence for gfx1250 (MI455X), wave32 + WMMA bf16.
//   B=256, T=256, D=128, H=384, L=12, C=32, G=4H+2L=1560 (padded -> 1568)
//
// Structure:
//   prep      : x->bf16, Wk/Wr -> pre-swizzled WMMA B fragments, fold biases
//   xproj     : Xp[b*T+t,:] = [x,t]@Wk + bk + br + t*(Wk[D,:]+Wr[H,:])  (WMMA)
//   fused x255: gates(step s-1) in-block -> h in LDS -> WMMA h@Wr + Xp_s
//   final     : gates(step 255)
// ---------------------------------------------------------------------------

#define Bb   256
#define Tt   256
#define Dd   128
#define Hh   384
#define Ll   12
#define Gg   1560
#define Gp   1568          // padded to 98 tiles of 16
#define NT   98            // N tiles
#define KC_X 4             // k-chunks of 32 for K=128
#define KC_H 12            // k-chunks of 32 for K=384

typedef __attribute__((ext_vector_type(16))) __bf16 v16bf;
typedef __attribute__((ext_vector_type(8)))  float  v8f;

union Frag {           // one WMMA 16-bit operand: 8 VGPRs / lane
    v16bf v;
    uint4 q[2];
};

__device__ __forceinline__ unsigned short bf16b(float f) {
    unsigned u = __float_as_uint(f);
    u += 0x7FFFu + ((u >> 16) & 1u);          // round-to-nearest-even
    return (unsigned short)(u >> 16);
}

__device__ __forceinline__ float sigf(float x) {
    return 1.f / (1.f + __expf(-x));
}

// ---------------------------------------------------------------------------
// Prep: x -> bf16
// ---------------------------------------------------------------------------
__global__ __launch_bounds__(256) void pack_x_bf16(
    const float* __restrict__ x, unsigned short* __restrict__ xB, int n)
{
    int i = blockIdx.x * 256 + threadIdx.x;
    if (i < n) xB[i] = bf16b(x[i]);
}

// ---------------------------------------------------------------------------
// Prep: swizzle Wk[:128,:] and Wr[:384,:] into per-lane WMMA B fragments.
// B-fragment layout (16-bit, 32x16): VGPR j, lanes 0-15: K=2j,2j+1 ;
// lanes 16-31: K=16+2j,16+2j+1 ; N = lane&15.
// Frag dword index: ((tileN*KC + kc)*32 + lane)*8 + j
// ---------------------------------------------------------------------------
__global__ __launch_bounds__(256) void pack_wfrag(
    const float* __restrict__ Wk, const float* __restrict__ Wr,
    unsigned int* __restrict__ WkF, unsigned int* __restrict__ WrF)
{
    int id = blockIdx.x * 256 + threadIdx.x;
    const int NWK = NT * KC_X * 32 * 8;       // 100352
    const int NWR = NT * KC_H * 32 * 8;       // 301056
    if (id < NWK) {
        int j = id & 7, lane = (id >> 3) & 31, kc = (id >> 8) & 3, tn = id >> 10;
        int n = tn * 16 + (lane & 15), half = lane >> 4;
        int k0 = kc * 32 + half * 16 + j * 2;
        unsigned lo = 0, hi = 0;
        if (n < Gg) {
            lo = bf16b(Wk[(size_t)k0       * Gg + n]);
            hi = bf16b(Wk[(size_t)(k0 + 1) * Gg + n]);
        }
        WkF[id] = lo | (hi << 16);
        return;
    }
    id -= NWK;
    if (id < NWR) {
        int j = id & 7, lane = (id >> 3) & 31, kc = (id >> 8) % 12, tn = id / 3072;
        int n = tn * 16 + (lane & 15), half = lane >> 4;
        int k0 = kc * 32 + half * 16 + j * 2;
        unsigned lo = 0, hi = 0;
        if (n < Gg) {
            lo = bf16b(Wr[(size_t)k0       * Gg + n]);
            hi = bf16b(Wr[(size_t)(k0 + 1) * Gg + n]);
        }
        WrF[id] = lo | (hi << 16);
    }
}

// ---------------------------------------------------------------------------
// Prep: TC[g] = Wk[D,g] + Wr[H,g]  (time-column rank-1 term, folded)
//       SC[g] = bk[g] + br[g]      (biases, folded)
//       zero c state buffer 0      -- every call (graph replay determinism)
// ---------------------------------------------------------------------------
__global__ __launch_bounds__(256) void pack_const_zero(
    const float* __restrict__ Wk, const float* __restrict__ Wr,
    const float* __restrict__ bk, const float* __restrict__ br,
    float* __restrict__ TC, float* __restrict__ SC,
    float* __restrict__ c0)
{
    int i = blockIdx.x * 256 + threadIdx.x;
    if (i < Gp) {
        if (i < Gg) {
            TC[i] = Wk[(size_t)Dd * Gg + i] + Wr[(size_t)Hh * Gg + i];
            SC[i] = bk[i] + br[i];
        } else { TC[i] = 0.f; SC[i] = 0.f; }
    }
    if (i < Bb * Hh) c0[i] = 0.f;
}

// ---------------------------------------------------------------------------
// Phase 1: Xp[r,g] = xB[r,:] @ WkF[:,g] + time[r]*TC[g] + SC[g]
// r = b*T + t over all 65536 rows.  One wave per 16x16 tile, K=128 (4 chunks).
// A layout (16-bit, 16x32): lane half 0: regs0-3 K 0-7, regs4-7 K 8-15;
//                           lane half 1: regs0-3 K 16-23, regs4-7 K 24-31.
// ---------------------------------------------------------------------------
__global__ __launch_bounds__(256) void xproj_gemm(
    const uint4* __restrict__ xB,          // 65536 rows x 16 uint4 (128 bf16)
    const uint4* __restrict__ WkF,
    const float* __restrict__ timev,       // B*T
    const float* __restrict__ TC, const float* __restrict__ SC,
    float* __restrict__ Xp)                // 65536 x Gp
{
    int wave = (blockIdx.x * 256 + (int)threadIdx.x) >> 5;
    int lane = threadIdx.x & 31;
    int tm = wave / NT, tn = wave % NT;
    int half = lane >> 4, nl = lane & 15;
    int col = tn * 16 + nl;

    float tcc = TC[col], scc = SC[col];
    int rbase = tm * 16 + 8 * half;        // C/D layout: row = v + 8*half
    v8f acc;
#pragma unroll
    for (int v = 0; v < 8; ++v) acc[v] = timev[rbase + v] * tcc + scc;

    int m = tm * 16 + nl;
    const uint4* xr = xB + (size_t)m * 16;
#pragma unroll
    for (int kc = 0; kc < KC_X; ++kc) {
        Frag a, b;
        a.q[0] = xr[kc * 4 + 2 * half];
        a.q[1] = xr[kc * 4 + 2 * half + 1];
        const uint4* bp = WkF + ((size_t)(tn * KC_X + kc) * 32 + lane) * 2;
        b.q[0] = bp[0];
        b.q[1] = bp[1];
        acc = __builtin_amdgcn_wmma_f32_16x16x32_bf16(
                  false, a.v, false, b.v, (short)0, acc, false, false);
    }
    float* op = Xp + (size_t)rbase * Gp + col;
#pragma unroll
    for (int v = 0; v < 8; ++v) op[(size_t)v * Gp] = acc[v];
}

// ---------------------------------------------------------------------------
// Fused recurrent step s (s = 1..T-1):
//   gates for step s-1 over this block's 16 batch rows -> h_{s-1} bf16 in LDS
//   xout_s = Xp_s + h_{s-1} @ Wr      (WMMA, A fragments fed from LDS)
// grid (tm=16, nb=4), 1024 threads (32 waves); wave w handles tn = nb*32+w.
// Block nb==0 is the unique writer of c_{s-1} (double-buffered) and d_out.
// ---------------------------------------------------------------------------
__global__ __launch_bounds__(1024) void step_fused(
    const float* __restrict__ prev,        // activations of step s-1, row base
    int prevStride,                        // Gp, or T*Gp when prev==Xp (s==1)
    const float* __restrict__ cread,       // c_{s-2}
    float* __restrict__ cwrite,            // c_{s-1}  (nb==0 writes)
    const uint4* __restrict__ WrF,
    const float* __restrict__ Xp_s,        // Xp + s*Gp ; batch stride T*Gp
    float* __restrict__ xout,              // xbuf[s&1] : B x Gp
    float* __restrict__ out, int tprev)    // h_{s-1} -> out[:, tprev, :]
{
    __shared__ float zsh[16][2 * Ll];
    __shared__ float fmA[16][Ll], imA[16][Ll];
    __shared__ __align__(16) unsigned short hL[16][Hh];   // h_{s-1} bf16

    const int tm = blockIdx.x, nb = blockIdx.y;
    const int tid = threadIdx.x;
    const int wave = tid >> 5, lane = tid & 31;
    const int tn = nb * 32 + wave;
    const int half = lane >> 4, nl = lane & 15;
    const bool active = (tn < NT);
    const int col = tn * 16 + nl;

    // prefetch WMMA C operand (Xp_s slice) so its latency overlaps gate math
    v8f acc;
    if (active) {
        const float* xp = Xp_s + (size_t)(tm * 16 + 8 * half) * Tt * Gp + col;
#pragma unroll
        for (int v = 0; v < 8; ++v) acc[v] = xp[(size_t)v * Tt * Gp];
    }

    // ---- master-gate logits into LDS ----
    if (tid < 16 * 2 * Ll) {
        int rr = tid / (2 * Ll), j = tid % (2 * Ll);
        zsh[rr][j] = prev[(size_t)(tm * 16 + rr) * prevStride + j];
    }
    __syncthreads();

    // ---- cumax l2r / r2l per (row, level) ----
    if (tid < 16 * Ll) {
        int rr = tid / Ll, l = tid % Ll;
        float m1 = zsh[rr][0], m2 = zsh[rr][Ll];
#pragma unroll
        for (int j = 1; j < Ll; ++j) {
            m1 = fmaxf(m1, zsh[rr][j]);
            m2 = fmaxf(m2, zsh[rr][Ll + j]);
        }
        float t1 = 0.f, c1 = 0.f, t2 = 0.f, c2 = 0.f;
#pragma unroll
        for (int j = 0; j < Ll; ++j) {
            float e1 = __expf(zsh[rr][j] - m1);       t1 += e1; if (j <= l) c1 += e1;
            float e2 = __expf(zsh[rr][Ll + j] - m2);  t2 += e2; if (j >= l) c2 += e2;
        }
        fmA[rr][l] = c1 / t1;
        imA[rr][l] = c2 / t2;
    }
    __syncthreads();

    // ---- cell update: 16 rows x 384, 64 threads/row, 6 elems/thread ----
    {
        int rr = tid >> 6, tc = tid & 63;
        int b  = tm * 16 + rr;
        const float* prow = prev + (size_t)b * prevStride;
#pragma unroll
        for (int e = 0; e < 6; ++e) {
            int h = tc + 64 * e;
            int l = h >> 5;
            float fm = fmA[rr][l], im = imA[rr][l];
            float zf = prow[2 * Ll + h];
            float zi = prow[2 * Ll + Hh + h];
            float zo = prow[2 * Ll + 2 * Hh + h];
            float zc = prow[2 * Ll + 3 * Hh + h];
            float fg = sigf(zf), ig = sigf(zi), og = sigf(zo);
            float ci = tanhf(zc);
            float cl = cread[(size_t)b * Hh + h];
            float ov = fm * im;
            float co = ov * (fg * cl + ig * ci) + (fm - ov) * cl + (im - ov) * ci;
            float ho = og * tanhf(co);
            hL[rr][h] = bf16b(ho);
            if (nb == 0) {
                cwrite[(size_t)b * Hh + h] = co;
                out[((size_t)b * Tt + tprev) * Hh + h] = ho;
            }
        }
    }
    __syncthreads();

    if (!active) return;

    // ---- WMMA: xout_s = Xp_s + h_{s-1} @ Wr ; A fragments from LDS ----
    const uint4* hr = (const uint4*)&hL[nl][0];   // 48 uint4 per row
#pragma unroll
    for (int kc = 0; kc < KC_H; ++kc) {
        Frag a, b;
        a.q[0] = hr[kc * 4 + 2 * half];
        a.q[1] = hr[kc * 4 + 2 * half + 1];
        const uint4* bp = WrF + ((size_t)(tn * KC_H + kc) * 32 + lane) * 2;
        b.q[0] = bp[0];
        b.q[1] = bp[1];
        acc = __builtin_amdgcn_wmma_f32_16x16x32_bf16(
                  false, a.v, false, b.v, (short)0, acc, false, false);
    }
    float* op = xout + (size_t)(tm * 16 + 8 * half) * Gp + col;
#pragma unroll
    for (int v = 0; v < 8; ++v) op[(size_t)v * Gp] = acc[v];
}

// ---------------------------------------------------------------------------
// Final gates (step T-1): emits h_{T-1} only.
// ---------------------------------------------------------------------------
__global__ __launch_bounds__(384) void gates_final(
    const float* __restrict__ xout,        // B x Gp (xbuf[(T-1)&1])
    const float* __restrict__ cread,       // c_{T-2}
    float* __restrict__ out, int t)
{
    __shared__ float z[2 * Ll];
    int b = blockIdx.x, h = threadIdx.x;
    const float* xr = xout + (size_t)b * Gp;
    if (h < 2 * Ll) z[h] = xr[h];
    __syncthreads();

    int l = h >> 5;
    float m1 = z[0], m2 = z[Ll];
#pragma unroll
    for (int j = 1; j < Ll; ++j) {
        m1 = fmaxf(m1, z[j]);
        m2 = fmaxf(m2, z[Ll + j]);
    }
    float t1 = 0.f, c1 = 0.f, t2 = 0.f, c2 = 0.f;
#pragma unroll
    for (int j = 0; j < Ll; ++j) {
        float e1 = __expf(z[j] - m1);       t1 += e1; if (j <= l) c1 += e1;
        float e2 = __expf(z[Ll + j] - m2);  t2 += e2; if (j >= l) c2 += e2;
    }
    float fm = c1 / t1, im = c2 / t2;

    float zf = xr[2 * Ll + h];
    float zi = xr[2 * Ll + Hh + h];
    float zo = xr[2 * Ll + 2 * Hh + h];
    float zc = xr[2 * Ll + 3 * Hh + h];
    float fg = sigf(zf), ig = sigf(zi), og = sigf(zo);
    float ci = tanhf(zc);
    float cl = cread[(size_t)b * Hh + h];
    float ov = fm * im;
    float co = ov * (fg * cl + ig * ci) + (fm - ov) * cl + (im - ov) * ci;
    out[((size_t)b * Tt + t) * Hh + h] = og * tanhf(co);
}

// ---------------------------------------------------------------------------
// Host-side orchestration
// ---------------------------------------------------------------------------
extern "C" void kernel_launch(void* const* d_in, const int* in_sizes, int n_in,
                              void* d_out, int out_size, void* d_ws, size_t ws_size,
                              hipStream_t stream)
{
    const float* x    = (const float*)d_in[0];   // (B,T,D)
    const float* timv = (const float*)d_in[1];   // (B,T)
    const float* Wk   = (const float*)d_in[2];   // (D+1,G)
    const float* bk   = (const float*)d_in[3];   // (G)
    const float* Wr   = (const float*)d_in[4];   // (H+1,G)
    const float* br   = (const float*)d_in[5];   // (G)
    float* out        = (float*)d_out;           // (B,T,H)

    // workspace carve-up (all offsets 16B-aligned)
    char* ws = (char*)d_ws;
    const size_t SZ_XP   = (size_t)Bb * Tt * Gp * 4;     // 411,041,792
    const size_t SZ_XB   = (size_t)Bb * Tt * Dd * 2;     //  16,777,216
    const size_t SZ_WKF  = (size_t)NT * KC_X * 32 * 8 * 4;
    const size_t SZ_WRF  = (size_t)NT * KC_H * 32 * 8 * 4;
    const size_t SZ_TC   = (size_t)Gp * 4;
    const size_t SZ_SC   = (size_t)Gp * 4;
    const size_t SZ_C    = (size_t)Bb * Hh * 4;
    const size_t SZ_XOUT = (size_t)Bb * Gp * 4;

    float*          Xp    = (float*)ws;                        ws += SZ_XP;
    unsigned short* xB    = (unsigned short*)ws;               ws += SZ_XB;
    unsigned int*   WkF   = (unsigned int*)ws;                 ws += SZ_WKF;
    unsigned int*   WrF   = (unsigned int*)ws;                 ws += SZ_WRF;
    float*          TC    = (float*)ws;                        ws += SZ_TC;
    float*          SC    = (float*)ws;                        ws += SZ_SC;
    float*          cb0   = (float*)ws;                        ws += SZ_C;
    float*          cb1   = (float*)ws;                        ws += SZ_C;
    float*          xb0   = (float*)ws;                        ws += SZ_XOUT;
    float*          xb1   = (float*)ws;
    float* cb[2]   = { cb0, cb1 };
    float* xbuf[2] = { xb0, xb1 };

    // --- prep ---
    {
        int n = Bb * Tt * Dd;
        pack_x_bf16<<<(n + 255) / 256, 256, 0, stream>>>(x, xB, n);
    }
    {
        int total = NT * KC_X * 32 * 8 + NT * KC_H * 32 * 8;   // 401,408
        pack_wfrag<<<(total + 255) / 256, 256, 0, stream>>>(Wk, Wr, WkF, WrF);
    }
    pack_const_zero<<<(Bb * Hh + 255) / 256, 256, 0, stream>>>(
        Wk, Wr, bk, br, TC, SC, cb0);

    // --- phase 1: all-timestep X-projection, 4096 x 98 tiles ---
    {
        int waves  = (Bb * Tt / 16) * NT;        // 401,408
        int blocks = waves / 8;                  //  50,176  (exact)
        xproj_gemm<<<blocks, 256, 0, stream>>>(
            (const uint4*)xB, (const uint4*)WkF, timv, TC, SC, Xp);
    }

    // --- phase 2: sequential recurrence (one fused kernel per step) ---
    // kernel s: gates for step s-1 (reads prev activations + c_{s-2}),
    //           then xout_s = Xp_s + h_{s-1} @ Wr.
    for (int s = 1; s < Tt; ++s) {
        const float* prev = (s == 1) ? Xp : xbuf[(s - 1) & 1];
        int pstr          = (s == 1) ? Tt * Gp : Gp;
        step_fused<<<dim3(16, 4), 1024, 0, stream>>>(
            prev, pstr, cb[(s - 1) & 1], cb[s & 1],
            (const uint4*)WrF, Xp + (size_t)s * Gp, xbuf[s & 1], out, s - 1);
    }
    gates_final<<<Bb, Hh, 0, stream>>>(
        xbuf[(Tt - 1) & 1], cb[(Tt - 1) & 1], out, Tt - 1);
}
